// ClinicalLongformerPool_25013889532531
// MI455X (gfx1250) — matrix operand order repaired and checked by reading, not compiled, not run on hardware
//
#include <hip/hip_runtime.h>
#include <math.h>

// ---------------------------------------------------------------------------
// Clinical Longformer encoder forward for gfx1250 (CDNA5, wave32, WMMA).
// All GEMMs + attention matmuls use v_wmma_f32_16x16x32_bf16.
// ---------------------------------------------------------------------------

typedef unsigned short u16;
typedef unsigned int   u32;
typedef __attribute__((ext_vector_type(16))) __bf16 v16bf;
typedef __attribute__((ext_vector_type(8)))  float  v8f;

struct __attribute__((aligned(16))) U4 { u32 x, y, z, w; };
union FragU { U4 q[2]; v16bf v; u16 u[16]; };

constexpr int Bb = 2, Ss = 4096, Dd = 768, Hh = 12, DHd = 64, Ll = 2;
constexpr int Cc = 256, NCHn = 16, Gg = 32, Ww = 256, FFf = 3072, NLn = 25;
constexpr float NEGV = -1.0e9f;

#define DEVFN __device__ __forceinline__

DEVFN u16 f2bf(float f) {                      // f32 -> bf16 round-to-nearest-even
  u32 u = __float_as_uint(f);
  u32 r = (u + 0x7fffu + ((u >> 16) & 1u)) >> 16;
  return (u16)r;
}

DEVFN v16bf load_frag2(const u16* p0, const u16* p1) {   // two 16B pieces -> frag
  FragU f;
  f.q[0] = *(const U4*)p0;
  f.q[1] = *(const U4*)p1;
  return f.v;
}

DEVFN v8f wmma_bf(v16bf a, v16bf b, v8f c) {
  return __builtin_amdgcn_wmma_f32_16x16x32_bf16(false, a, false, b, (short)0, c,
                                                 false, false);
}

DEVFN float red16_max(float v) {               // reduce over 16-lane half-group
#pragma unroll
  for (int m = 1; m < 16; m <<= 1) v = fmaxf(v, __shfl_xor(v, m, 32));
  return v;
}
DEVFN float red16_sum(float v) {
#pragma unroll
  for (int m = 1; m < 16; m <<= 1) v += __shfl_xor(v, m, 32);
  return v;
}

DEVFN float gelu_f(float x) {                  // exact gelu (approximate=False)
  return 0.5f * x * (1.0f + erff(x * 0.70710678118654752f));
}

DEVFN float block_sum256(float v, float* sbuf) {
#pragma unroll
  for (int m = 1; m < 32; m <<= 1) v += __shfl_xor(v, m, 32);
  int w = threadIdx.x >> 5;
  if ((threadIdx.x & 31) == 0) sbuf[w] = v;
  __syncthreads();
  float t = 0.f;
#pragma unroll
  for (int i = 0; i < 8; ++i) t += sbuf[i];
  __syncthreads();
  return t;
}

// ---------------------------------------------------------------------------
// f32 -> bf16 bulk convert (weights)
// ---------------------------------------------------------------------------
__global__ __launch_bounds__(256) void cvt_bf16_kernel(const float* __restrict__ in,
                                                       u16* __restrict__ out, int n) {
  int i = blockIdx.x * 256 + threadIdx.x;
  if (i < n) out[i] = f2bf(in[i]);
}

// ---------------------------------------------------------------------------
// embedding + layernorm : x = LN(word_emb[id] + pos_emb[s]) ; one block per token
// ---------------------------------------------------------------------------
__global__ __launch_bounds__(256)
void embed_ln_kernel(const int* __restrict__ ids, const float* __restrict__ wemb,
                     const float* __restrict__ pemb, const float* __restrict__ gs,
                     const float* __restrict__ gb, float* xf, u16* xb) {
  __shared__ float sbuf[8];
  int row = blockIdx.x;
  int s = row % Ss;
  int id = ids[row];
  float vals[3];
  float sum = 0.f;
#pragma unroll
  for (int i = 0; i < 3; ++i) {
    int d = threadIdx.x + i * 256;
    float vv = wemb[(size_t)id * Dd + d] + pemb[(size_t)s * Dd + d];
    vals[i] = vv; sum += vv;
  }
  float mu = block_sum256(sum, sbuf) * (1.0f / Dd);
  float vs = 0.f;
#pragma unroll
  for (int i = 0; i < 3; ++i) { float dv = vals[i] - mu; vs += dv * dv; }
  float var = block_sum256(vs, sbuf) * (1.0f / Dd);
  float rstd = rsqrtf(var + 1e-5f);
#pragma unroll
  for (int i = 0; i < 3; ++i) {
    int d = threadIdx.x + i * 256;
    float y = (vals[i] - mu) * rstd * gs[d] + gb[d];
    xf[(size_t)row * Dd + d] = y;
    xb[(size_t)row * Dd + d] = f2bf(y);
  }
}

// ---------------------------------------------------------------------------
// residual add + layernorm : x = LN(x + y)
// ---------------------------------------------------------------------------
__global__ __launch_bounds__(256)
void add_ln_kernel(const float* xin, const float* yin, const float* __restrict__ gs,
                   const float* __restrict__ gb, float* xof, u16* xob) {
  __shared__ float sbuf[8];
  int row = blockIdx.x;
  float vals[3];
  float sum = 0.f;
#pragma unroll
  for (int i = 0; i < 3; ++i) {
    int d = threadIdx.x + i * 256;
    float vv = xin[(size_t)row * Dd + d] + yin[(size_t)row * Dd + d];
    vals[i] = vv; sum += vv;
  }
  float mu = block_sum256(sum, sbuf) * (1.0f / Dd);
  float vs = 0.f;
#pragma unroll
  for (int i = 0; i < 3; ++i) { float dv = vals[i] - mu; vs += dv * dv; }
  float var = block_sum256(vs, sbuf) * (1.0f / Dd);
  float rstd = rsqrtf(var + 1e-5f);
#pragma unroll
  for (int i = 0; i < 3; ++i) {
    int d = threadIdx.x + i * 256;
    float y = (vals[i] - mu) * rstd * gs[d] + gb[d];
    xof[(size_t)row * Dd + d] = y;
    xob[(size_t)row * Dd + d] = f2bf(y);
  }
}

// ---------------------------------------------------------------------------
// Generic WMMA GEMM: C = act((A @ W + bias) * scale)
//   A : [M,K] bf16 row-major      W : [K,N] bf16 row-major
//   block tile 128x64, 8 waves of 32x32, K-step 32.
// ---------------------------------------------------------------------------
template <int ACT>
__global__ __launch_bounds__(256)
void gemm_kernel(const u16* __restrict__ A, const u16* __restrict__ Bw,
                 const float* __restrict__ bias, float* Cf, u16* Cb,
                 int M, int N, int K, float scale) {
  __shared__ u16 As[128][40];   // [row][k] pad->80B rows (16B aligned)
  __shared__ u16 Bs[64][40];    // [n][k]   (W tile pre-transposed)
  const int tid = threadIdx.x;
  const int lane = tid & 31, wave = tid >> 5;
  const int hl = lane >> 4, l16 = lane & 15;
  const int m0 = blockIdx.y * 128, n0 = blockIdx.x * 64;
  const int wm = (wave & 3) * 32, wn = (wave >> 2) * 32;

  v8f acc[2][2] = {};

  for (int k0 = 0; k0 < K; k0 += 32) {
    // A tile: 128x32 via 8-bf16 (16B) pieces
    for (int idx = tid; idx < 512; idx += 256) {
      int r = idx >> 2, seg = (idx & 3) << 3;
      U4 val = {0, 0, 0, 0};
      int gr = m0 + r;
      if (gr < M) val = *(const U4*)(A + (size_t)gr * K + k0 + seg);
      *(U4*)(&As[r][seg]) = val;
    }
    // B tile transposed: Bs[n][kk] = W[k0+kk][n0+n]
    for (int idx = tid; idx < 2048; idx += 256) {
      int nn = idx & 63, kk = idx >> 6;
      Bs[nn][kk] = Bw[(size_t)(k0 + kk) * N + n0 + nn];
    }
    // prefetch next A tile
    if (k0 + 32 < K) {
      int pr = m0 + (tid >> 1);
      if (pr < M) __builtin_prefetch(A + (size_t)pr * K + k0 + 32, 0, 1);
    }
    __syncthreads();

    v16bf af[2], bfr[2];
#pragma unroll
    for (int ms = 0; ms < 2; ++ms) {
      const u16* p = &As[wm + ms * 16 + l16][0];
      af[ms] = load_frag2(p + hl * 16, p + hl * 16 + 8);
    }
#pragma unroll
    for (int ns = 0; ns < 2; ++ns) {
      const u16* p = &Bs[wn + ns * 16 + l16][0];
      bfr[ns] = load_frag2(p + hl * 16, p + hl * 16 + 8);
    }
#pragma unroll
    for (int ms = 0; ms < 2; ++ms)
#pragma unroll
      for (int ns = 0; ns < 2; ++ns)
        acc[ms][ns] = wmma_bf(af[ms], bfr[ns], acc[ms][ns]);
    __syncthreads();
  }

  for (int ms = 0; ms < 2; ++ms)
    for (int ns = 0; ns < 2; ++ns) {
      int col = n0 + wn + ns * 16 + l16;
      float bv = bias[col];
#pragma unroll
      for (int r = 0; r < 8; ++r) {
        int row = m0 + wm + ms * 16 + r + 8 * hl;
        if (row < M) {
          float v = (acc[ms][ns][r] + bv) * scale;
          if (ACT == 1) v = gelu_f(v);
          size_t o = (size_t)row * N + col;
          if (Cf) Cf[o] = v;
          if (Cb) Cb[o] = f2bf(v);
        }
      }
    }
}

// ---------------------------------------------------------------------------
// gather x[:, :G] rows into compact [B*G, D] bf16 buffer
// ---------------------------------------------------------------------------
__global__ void gather_xg_kernel(const u16* __restrict__ xb, u16* __restrict__ xg) {
  int i = blockIdx.x * 256 + threadIdx.x;
  if (i < Bb * Gg * Dd) {
    int d = i % Dd; int r = i / Dd; int g = r % Gg; int b = r / Gg;
    xg[i] = xb[((size_t)(b * Ss + g)) * Dd + d];
  }
}

// ---------------------------------------------------------------------------
// Local (sliding-window + global-key) attention, flash-style online softmax.
// One block = (b, h, chunk); 8 waves, each wave = 32 query rows.
// Key blocks: kb==0 -> 32 global keys (cols 32..63 masked); kb>0 -> 64 banded keys.
// Writes bf16 context [B,S,D] (head h -> cols h*64..h*64+63).
// ---------------------------------------------------------------------------
__global__ __launch_bounds__(256)
void local_attn_kernel(const u16* __restrict__ q, const u16* __restrict__ k,
                       const u16* __restrict__ v, const int* __restrict__ mask,
                       u16* __restrict__ ctx) {
  __shared__ u16 kt[64][72];       // [key][d]
  __shared__ u16 vt[64][72];       // [d][key] (transposed V)
  __shared__ u16 ps[8][32][64];    // per-wave P staging (C-layout -> A-layout)
  const int bid = blockIdx.x;
  const int n = bid % NCHn, h = (bid / NCHn) % Hh, b = bid / (NCHn * Hh);
  const int tid = threadIdx.x;
  const int lane = tid & 31, wave = tid >> 5, hl = lane >> 4, l16 = lane & 15;
  const int qr0 = wave * 32;

  // Q fragments held in registers for the whole key loop (already scaled).
  v16bf qa[2][2];
#pragma unroll
  for (int ms = 0; ms < 2; ++ms) {
    int srow = n * Cc + qr0 + ms * 16 + l16;
    const u16* base = q + ((size_t)(b * Ss + srow)) * Dd + h * DHd;
#pragma unroll
    for (int ks = 0; ks < 2; ++ks)
      qa[ms][ks] = load_frag2(base + ks * 32 + hl * 16, base + ks * 32 + hl * 16 + 8);
  }

  v8f oacc[2][4] = {};
  float rmx[2][8], rsm[2][8];
#pragma unroll
  for (int ms = 0; ms < 2; ++ms)
#pragma unroll
    for (int r = 0; r < 8; ++r) { rmx[ms][r] = -INFINITY; rsm[ms][r] = 0.f; }

  for (int kb = 0; kb < 13; ++kb) {
    const bool isglob = (kb == 0);
    const int kbase = isglob ? 0 : (n * Cc + (kb - 1) * 64 - Cc);
    if (!isglob && (kbase + 63 < Gg || kbase >= Ss)) continue;  // uniform skip

    // load K block [key][d] (16B pieces) and V block transposed [d][key]
    for (int idx = tid; idx < 512; idx += 256) {
      int key = idx >> 3, seg = (idx & 7) << 3;
      int kidx = kbase + key;
      U4 val = {0, 0, 0, 0};
      if (kidx >= 0 && kidx < Ss && !(isglob && key >= Gg))
        val = *(const U4*)(k + ((size_t)(b * Ss + kidx)) * Dd + h * DHd + seg);
      *(U4*)(&kt[key][seg]) = val;
    }
    for (int idx = tid; idx < 4096; idx += 256) {
      int d = idx & 63, key = idx >> 6;
      int kidx = kbase + key;
      u16 val = 0;
      if (kidx >= 0 && kidx < Ss && !(isglob && key >= Gg))
        val = v[((size_t)(b * Ss + kidx)) * Dd + h * DHd + d];
      vt[d][key] = val;
    }
    __syncthreads();

    // scores: S = Q @ K^T  (2x4 fragments of 16x16)
    v8f sc[2][4];
#pragma unroll
    for (int ns = 0; ns < 4; ++ns) {
      const u16* kp = &kt[ns * 16 + l16][0];
      v16bf b0 = load_frag2(kp + hl * 16, kp + hl * 16 + 8);
      v16bf b1 = load_frag2(kp + 32 + hl * 16, kp + 32 + hl * 16 + 8);
#pragma unroll
      for (int ms = 0; ms < 2; ++ms) {
        v8f c = {};
        c = wmma_bf(qa[ms][0], b0, c);
        c = wmma_bf(qa[ms][1], b1, c);
        sc[ms][ns] = c;
      }
    }

    // mask + online softmax per 16-row group
#pragma unroll
    for (int ms = 0; ms < 2; ++ms) {
      float nmax[8];
#pragma unroll
      for (int r = 0; r < 8; ++r) nmax[r] = rmx[ms][r];
#pragma unroll
      for (int ns = 0; ns < 4; ++ns) {
        int kcol = ns * 16 + l16;
        int kidx = kbase + kcol;
        bool kval;
        if (isglob) kval = (kcol < Gg) && (mask[b * Ss + kcol] != 0);
        else        kval = (kidx >= Gg) && (kidx < Ss) && (mask[b * Ss + kidx] != 0);
#pragma unroll
        for (int r = 0; r < 8; ++r) {
          int sq = n * Cc + qr0 + ms * 16 + r + 8 * hl;
          bool ok = kval;
          if (!isglob) { int dd = kidx - sq; ok = ok && (dd >= -Ww) && (dd <= Ww); }
          float val = ok ? sc[ms][ns][r] : NEGV;
          sc[ms][ns][r] = val;
          nmax[r] = fmaxf(nmax[r], val);
        }
      }
      float rsum[8];
#pragma unroll
      for (int r = 0; r < 8; ++r) { nmax[r] = red16_max(nmax[r]); rsum[r] = 0.f; }
#pragma unroll
      for (int ns = 0; ns < 4; ++ns)
#pragma unroll
        for (int r = 0; r < 8; ++r) {
          float p = __expf(sc[ms][ns][r] - nmax[r]);
          sc[ms][ns][r] = p;
          rsum[r] += p;
        }
#pragma unroll
      for (int r = 0; r < 8; ++r) {
        rsum[r] = red16_sum(rsum[r]);
        float alpha = __expf(rmx[ms][r] - nmax[r]);
        rmx[ms][r] = nmax[r];
        rsm[ms][r] = rsm[ms][r] * alpha + rsum[r];
#pragma unroll
        for (int ds = 0; ds < 4; ++ds) oacc[ms][ds][r] *= alpha;
      }
      // stage P (bf16) so it can be reloaded in the A-fragment layout
#pragma unroll
      for (int ns = 0; ns < 4; ++ns)
#pragma unroll
        for (int r = 0; r < 8; ++r)
          ps[wave][ms * 16 + r + 8 * hl][ns * 16 + l16] = f2bf(sc[ms][ns][r]);
    }

    // O += P @ V
#pragma unroll
    for (int ks = 0; ks < 2; ++ks) {
      v16bf pa[2];
#pragma unroll
      for (int ms = 0; ms < 2; ++ms) {
        const u16* pp = &ps[wave][ms * 16 + l16][0];
        pa[ms] = load_frag2(pp + ks * 32 + hl * 16, pp + ks * 32 + hl * 16 + 8);
      }
#pragma unroll
      for (int ds = 0; ds < 4; ++ds) {
        const u16* vp = &vt[ds * 16 + l16][0];
        v16bf vb = load_frag2(vp + ks * 32 + hl * 16, vp + ks * 32 + hl * 16 + 8);
#pragma unroll
        for (int ms = 0; ms < 2; ++ms)
          oacc[ms][ds] = wmma_bf(pa[ms], vb, oacc[ms][ds]);
      }
    }
    __syncthreads();
  }

  // normalize + store
  for (int ms = 0; ms < 2; ++ms)
    for (int ds = 0; ds < 4; ++ds)
#pragma unroll
      for (int r = 0; r < 8; ++r) {
        int srow = n * Cc + qr0 + ms * 16 + r + 8 * hl;
        float o = oacc[ms][ds][r] / rsm[ms][r];
        ctx[((size_t)(b * Ss + srow)) * Dd + h * DHd + ds * 16 + l16] = f2bf(o);
      }
}

// ---------------------------------------------------------------------------
// Global-token attention: rows s < G attend over all S keys (kg/vg).
// One wave per (b,h); K fragments loaded directly from global (layout matches).
// Overwrites ctx rows s < G (launched after local_attn).
// ---------------------------------------------------------------------------
__global__ __launch_bounds__(256)
void global_attn_kernel(const u16* __restrict__ qg, const u16* __restrict__ kg,
                        const u16* __restrict__ vg, const int* __restrict__ mask,
                        u16* __restrict__ ctx) {
  __shared__ u16 ps[8][32][64];
  const int tid = threadIdx.x;
  const int lane = tid & 31, wave = tid >> 5, hl = lane >> 4, l16 = lane & 15;
  const int pair = blockIdx.x * 8 + wave;
  if (pair >= Bb * Hh) return;   // wave-uniform exit; kernel has no barriers
  const int b = pair / Hh, h = pair % Hh;

  v16bf qa[2][2];
#pragma unroll
  for (int ms = 0; ms < 2; ++ms) {
    const u16* base = qg + ((size_t)(b * Gg + ms * 16 + l16)) * Dd + h * DHd;
#pragma unroll
    for (int ks = 0; ks < 2; ++ks)
      qa[ms][ks] = load_frag2(base + ks * 32 + hl * 16, base + ks * 32 + hl * 16 + 8);
  }

  v8f oacc[2][4] = {};
  float rmx[2][8], rsm[2][8];
#pragma unroll
  for (int ms = 0; ms < 2; ++ms)
#pragma unroll
    for (int r = 0; r < 8; ++r) { rmx[ms][r] = -INFINITY; rsm[ms][r] = 0.f; }

  for (int kb = 0; kb < Ss / 64; ++kb) {
    const int kbase = kb * 64;
    v8f sc[2][4];
#pragma unroll
    for (int ns = 0; ns < 4; ++ns) {
      const u16* kp = kg + ((size_t)(b * Ss + kbase + ns * 16 + l16)) * Dd + h * DHd;
      v16bf b0 = load_frag2(kp + hl * 16, kp + hl * 16 + 8);
      v16bf b1 = load_frag2(kp + 32 + hl * 16, kp + 32 + hl * 16 + 8);
#pragma unroll
      for (int ms = 0; ms < 2; ++ms) {
        v8f c = {};
        c = wmma_bf(qa[ms][0], b0, c);
        c = wmma_bf(qa[ms][1], b1, c);
        sc[ms][ns] = c;
      }
    }

#pragma unroll
    for (int ms = 0; ms < 2; ++ms) {
      float nmax[8];
#pragma unroll
      for (int r = 0; r < 8; ++r) nmax[r] = rmx[ms][r];
#pragma unroll
      for (int ns = 0; ns < 4; ++ns) {
        bool kval = (mask[b * Ss + kbase + ns * 16 + l16] != 0);
#pragma unroll
        for (int r = 0; r < 8; ++r) {
          float val = kval ? sc[ms][ns][r] : NEGV;
          sc[ms][ns][r] = val;
          nmax[r] = fmaxf(nmax[r], val);
        }
      }
      float rsum[8];
#pragma unroll
      for (int r = 0; r < 8; ++r) { nmax[r] = red16_max(nmax[r]); rsum[r] = 0.f; }
#pragma unroll
      for (int ns = 0; ns < 4; ++ns)
#pragma unroll
        for (int r = 0; r < 8; ++r) {
          float p = __expf(sc[ms][ns][r] - nmax[r]);
          sc[ms][ns][r] = p;
          rsum[r] += p;
        }
#pragma unroll
      for (int r = 0; r < 8; ++r) {
        rsum[r] = red16_sum(rsum[r]);
        float alpha = __expf(rmx[ms][r] - nmax[r]);
        rmx[ms][r] = nmax[r];
        rsm[ms][r] = rsm[ms][r] * alpha + rsum[r];
#pragma unroll
        for (int ds = 0; ds < 4; ++ds) oacc[ms][ds][r] *= alpha;
      }
#pragma unroll
      for (int ns = 0; ns < 4; ++ns)
#pragma unroll
        for (int r = 0; r < 8; ++r)
          ps[wave][ms * 16 + r + 8 * hl][ns * 16 + l16] = f2bf(sc[ms][ns][r]);
    }

#pragma unroll
    for (int ks = 0; ks < 2; ++ks) {
      v16bf pa[2];
#pragma unroll
      for (int ms = 0; ms < 2; ++ms) {
        const u16* pp = &ps[wave][ms * 16 + l16][0];
        pa[ms] = load_frag2(pp + ks * 32 + hl * 16, pp + ks * 32 + hl * 16 + 8);
      }
#pragma unroll
      for (int ds = 0; ds < 4; ++ds) {
        FragU vf;   // gather transposed V fragment directly from global
#pragma unroll
        for (int i = 0; i < 16; ++i) {
          int key = kbase + ks * 32 + hl * 16 + i;
          vf.u[i] = vg[((size_t)(b * Ss + key)) * Dd + h * DHd + ds * 16 + l16];
        }
#pragma unroll
        for (int ms = 0; ms < 2; ++ms)
          oacc[ms][ds] = wmma_bf(pa[ms], vf.v, oacc[ms][ds]);
      }
    }
  }

  for (int ms = 0; ms < 2; ++ms)
    for (int ds = 0; ds < 4; ++ds)
#pragma unroll
      for (int r = 0; r < 8; ++r) {
        int g = ms * 16 + r + 8 * hl;   // < G
        float o = oacc[ms][ds][r] / rsm[ms][r];
        ctx[((size_t)(b * Ss + g)) * Dd + h * DHd + ds * 16 + l16] = f2bf(o);
      }
}

// ---------------------------------------------------------------------------
// masked mean pooling : pooled[b,d] = sum_s x*m / sum_s m
// ---------------------------------------------------------------------------
__global__ __launch_bounds__(128)
void pool_kernel(const float* __restrict__ xf, const int* __restrict__ mask,
                 float* __restrict__ pooled) {
  __shared__ float sb[128];
  int b = blockIdx.x, d = blockIdx.y * 128 + threadIdx.x;
  float mp = 0.f;
  for (int s = threadIdx.x; s < Ss; s += 128) mp += (float)mask[b * Ss + s];
  sb[threadIdx.x] = mp;
  __syncthreads();
  for (int st = 64; st > 0; st >>= 1) {
    if (threadIdx.x < st) sb[threadIdx.x] += sb[threadIdx.x + st];
    __syncthreads();
  }
  float msum = sb[0];
  float acc = 0.f;
  for (int s = 0; s < Ss; ++s) {
    float m = (float)mask[b * Ss + s];
    acc += m * xf[((size_t)(b * Ss + s)) * Dd + d];
  }
  pooled[b * Dd + d] = acc / msum;
}

// ---------------------------------------------------------------------------
// classifier + BCE-with-logits loss ; out[0..49]=logits, out[50]=loss
// ---------------------------------------------------------------------------
__global__ __launch_bounds__(64)
void clf_kernel(const float* __restrict__ pooled, const float* __restrict__ w,
                const float* __restrict__ cb, const float* __restrict__ labels,
                float* __restrict__ out) {
  __shared__ float ls[64];
  int t = threadIdx.x;
  float term = 0.f;
  if (t < Bb * NLn) {
    int b = t / NLn, j = t % NLn;
    float z = cb[j];
    for (int d = 0; d < Dd; ++d) z += pooled[b * Dd + d] * w[d * NLn + j];
    out[t] = z;
    float y = labels[t];
    term = fmaxf(z, 0.f) - z * y + log1pf(__expf(-fabsf(z)));
  }
  ls[t] = term;
  __syncthreads();
  if (t == 0) {
    float s = 0.f;
    for (int i = 0; i < Bb * NLn; ++i) s += ls[i];
    out[Bb * NLn] = s / (float)(Bb * NLn);
  }
}

// ---------------------------------------------------------------------------
// host launcher
// ---------------------------------------------------------------------------
extern "C" void kernel_launch(void* const* d_in, const int* in_sizes, int n_in,
                              void* d_out, int out_size, void* d_ws, size_t ws_size,
                              hipStream_t stream) {
  (void)in_sizes; (void)n_in; (void)out_size; (void)ws_size;
  const int*   ids   = (const int*)d_in[0];
  const int*   amask = (const int*)d_in[1];
  const float* labels = (const float*)d_in[2];
  const float* word_emb = (const float*)d_in[3];
  const float* pos_emb  = (const float*)d_in[4];
  const float* eln_s = (const float*)d_in[5];
  const float* eln_b = (const float*)d_in[6];
  const float* Wq = (const float*)d_in[7];   const float* bq = (const float*)d_in[8];
  const float* Wk = (const float*)d_in[9];   const float* bk = (const float*)d_in[10];
  const float* Wv = (const float*)d_in[11];  const float* bv = (const float*)d_in[12];
  const float* Wqg = (const float*)d_in[13]; const float* bqg = (const float*)d_in[14];
  const float* Wkg = (const float*)d_in[15]; const float* bkg = (const float*)d_in[16];
  const float* Wvg = (const float*)d_in[17]; const float* bvg = (const float*)d_in[18];
  const float* Wo = (const float*)d_in[19];  const float* bo = (const float*)d_in[20];
  const float* ln1s = (const float*)d_in[21]; const float* ln1b = (const float*)d_in[22];
  const float* Wf1 = (const float*)d_in[23]; const float* bf1 = (const float*)d_in[24];
  const float* Wf2 = (const float*)d_in[25]; const float* bf2 = (const float*)d_in[26];
  const float* ln2s = (const float*)d_in[27]; const float* ln2b = (const float*)d_in[28];
  const float* clfw = (const float*)d_in[29]; const float* clfb = (const float*)d_in[30];

  char* ws = (char*)d_ws;
  size_t off = 0;
  auto alloc = [&](size_t bytes) -> void* {
    void* p = ws + off;
    off = (off + bytes + 255) & ~(size_t)255;
    return p;
  };
  const size_t NX = (size_t)Bb * Ss * Dd;          // 6.29M
  float* xf   = (float*)alloc(NX * 4);
  u16*   xb   = (u16*)alloc(NX * 2);
  u16*   ctxb = (u16*)alloc(NX * 2);
  float* proj = (float*)alloc(NX * 4);
  u16*   hb   = (u16*)alloc((size_t)Bb * Ss * FFf * 2);
  u16*   qbuf = (u16*)alloc(NX * 2);
  u16*   kbuf = (u16*)alloc(NX * 2);
  u16*   vbuf = (u16*)alloc(NX * 2);
  u16*   kgb  = (u16*)alloc(NX * 2);
  u16*   vgb  = (u16*)alloc(NX * 2);
  u16*   xgb  = (u16*)alloc((size_t)Bb * Gg * Dd * 2);
  u16*   qgb  = (u16*)alloc((size_t)Bb * Gg * Dd * 2);
  float* pooled = (float*)alloc((size_t)Bb * Dd * 4);
  const size_t WDD = (size_t)Ll * Dd * Dd;
  const size_t WDF = (size_t)Ll * Dd * FFf;
  u16* wqb  = (u16*)alloc(WDD * 2);
  u16* wkb  = (u16*)alloc(WDD * 2);
  u16* wvb  = (u16*)alloc(WDD * 2);
  u16* wqgb = (u16*)alloc(WDD * 2);
  u16* wkgb = (u16*)alloc(WDD * 2);
  u16* wvgb = (u16*)alloc(WDD * 2);
  u16* wob  = (u16*)alloc(WDD * 2);
  u16* wf1b = (u16*)alloc(WDF * 2);
  u16* wf2b = (u16*)alloc(WDF * 2);

  auto cvt = [&](const float* src, u16* dst, size_t n) {
    int blocks = (int)((n + 255) / 256);
    cvt_bf16_kernel<<<blocks, 256, 0, stream>>>(src, dst, (int)n);
  };
  cvt(Wq, wqb, WDD);   cvt(Wk, wkb, WDD);   cvt(Wv, wvb, WDD);
  cvt(Wqg, wqgb, WDD); cvt(Wkg, wkgb, WDD); cvt(Wvg, wvgb, WDD);
  cvt(Wo, wob, WDD);   cvt(Wf1, wf1b, WDF); cvt(Wf2, wf2b, WDF);

  auto gemm = [&](const u16* A, const u16* W, const float* bias, float* Cf, u16* Cb,
                  int M, int N, int K, float scale, int act) {
    dim3 g(N / 64, (M + 127) / 128);
    if (act) gemm_kernel<1><<<g, 256, 0, stream>>>(A, W, bias, Cf, Cb, M, N, K, scale);
    else     gemm_kernel<0><<<g, 256, 0, stream>>>(A, W, bias, Cf, Cb, M, N, K, scale);
  };

  // embeddings + LN
  embed_ln_kernel<<<Bb * Ss, 256, 0, stream>>>(ids, word_emb, pos_emb, eln_s, eln_b,
                                               xf, xb);

  const float scl = 0.125f;   // 1/sqrt(DH)
  for (int l = 0; l < Ll; ++l) {
    const size_t wdd = (size_t)l * Dd * Dd;
    gemm(xb, wqb + wdd, bq + l * Dd, nullptr, qbuf, Bb * Ss, Dd, Dd, scl, 0);
    gemm(xb, wkb + wdd, bk + l * Dd, nullptr, kbuf, Bb * Ss, Dd, Dd, 1.f, 0);
    gemm(xb, wvb + wdd, bv + l * Dd, nullptr, vbuf, Bb * Ss, Dd, Dd, 1.f, 0);
    gemm(xb, wkgb + wdd, bkg + l * Dd, nullptr, kgb, Bb * Ss, Dd, Dd, 1.f, 0);
    gemm(xb, wvgb + wdd, bvg + l * Dd, nullptr, vgb, Bb * Ss, Dd, Dd, 1.f, 0);
    gather_xg_kernel<<<(Bb * Gg * Dd + 255) / 256, 256, 0, stream>>>(xb, xgb);
    gemm(xgb, wqgb + wdd, bqg + l * Dd, nullptr, qgb, Bb * Gg, Dd, Dd, scl, 0);

    local_attn_kernel<<<Bb * Hh * NCHn, 256, 0, stream>>>(qbuf, kbuf, vbuf, amask, ctxb);
    global_attn_kernel<<<(Bb * Hh + 7) / 8, 256, 0, stream>>>(qgb, kgb, vgb, amask, ctxb);

    gemm(ctxb, wob + wdd, bo + l * Dd, proj, nullptr, Bb * Ss, Dd, Dd, 1.f, 0);
    add_ln_kernel<<<Bb * Ss, 256, 0, stream>>>(xf, proj, ln1s + l * Dd, ln1b + l * Dd,
                                               xf, xb);
    gemm(xb, wf1b + (size_t)l * Dd * FFf, bf1 + l * FFf, nullptr, hb,
         Bb * Ss, FFf, Dd, 1.f, 1);               // fused exact GELU
    gemm(hb, wf2b + (size_t)l * FFf * Dd, bf2 + l * Dd, proj, nullptr,
         Bb * Ss, Dd, FFf, 1.f, 0);
    add_ln_kernel<<<Bb * Ss, 256, 0, stream>>>(xf, proj, ln2s + l * Dd, ln2b + l * Dd,
                                               xf, xb);
  }

  pool_kernel<<<dim3(Bb, Dd / 128), 128, 0, stream>>>(xf, amask, pooled);
  clf_kernel<<<1, 64, 0, stream>>>(pooled, clfw, clfb, labels, (float*)d_out);
}
